// FeatureSampling_14551349199023
// MI455X (gfx1250) — matrix operand order repaired and checked by reading, not compiled, not run on hardware
//
#include <hip/hip_runtime.h>
#include <hip/hip_bf16.h>

// ---------------------------------------------------------------------------
// MI455X (gfx1250, wave32) implementation.
// Convs are implicit GEMMs on v_wmma_f32_16x16x32_bf16 (fp32 accumulate).
// Block tile 64 px x 128 cout: 32 WMMAs per staged LDS tile (4 per wave).
// Fragments are LDS-contiguous per lane -> ds_load_b128 into register quads.
// ---------------------------------------------------------------------------

typedef __attribute__((ext_vector_type(16))) __bf16 v16bf;
typedef __attribute__((ext_vector_type(8)))  float  v8f;

__device__ __forceinline__ unsigned short f32_to_bf16_rne(float f) {
    unsigned u = __float_as_uint(f);
    unsigned r = u + 0x7FFFu + ((u >> 16) & 1u);   // round-to-nearest-even
    return (unsigned short)(r >> 16);
}

__device__ __forceinline__ unsigned pack_bf16x2(float lo, float hi) {
    return (unsigned)f32_to_bf16_rne(lo) | ((unsigned)f32_to_bf16_rne(hi) << 16);
}

// ---------------------------------------------------------------------------
// Pack fp32 weights [O=256][I=256][3][3] into bf16-pair layout TRANSPOSED:
//   K index k = tap*256 + c (tap = kh*3+kw), kp = k/2
//   wpT[o*1152 + kp] = { bf16(w[o][c][tap]) , bf16(w[o][c+1][tap]) }
// so a (cout, kp-run) fragment is contiguous in memory/LDS.
// ---------------------------------------------------------------------------
__global__ void pack_w_bf16(const float* __restrict__ w, unsigned* __restrict__ wpT) {
    int i = blockIdx.x * 256 + threadIdx.x;        // 256*1152 entries
    if (i >= 256 * 1152) return;
    int kp = i % 1152;
    int o  = i / 1152;
    int k0 = kp * 2;
    int tap = k0 >> 8;                              // 0..8
    int c0  = k0 & 255;                             // even
    float lo = w[(size_t)o * 2304 + (size_t)c0       * 9 + tap];
    float hi = w[(size_t)o * 2304 + (size_t)(c0 + 1) * 9 + tap];
    wpT[(size_t)o * 1152 + kp] = pack_bf16x2(lo, hi);
}

// ---------------------------------------------------------------------------
// 3x3 conv, pad=1, Cin=Cout=256, batch 8, square HxW (H in {128,64,32}).
// 256 threads = 8 waves. Block tile: 64 pixels (M) x 128 couts (N).
// Wave -> M tile (wave&3), N half (wave>>2) of 64 couts = 4 accumulators.
// K loop: 9 taps x 8 steps of 32 channels (K ordered tap*256+c).
// ---------------------------------------------------------------------------
__global__ void conv3x3_wmma_bf16(const float* __restrict__ in,
                                  const unsigned* __restrict__ wpT,  // bf16 pairs [256][1152]
                                  const float* __restrict__ bias,
                                  float* __restrict__ out,
                                  int H, int W) {
    __shared__ unsigned ldsA[64 * 20];    // [pixel][kp]   stride 20 (16B-aligned frag runs)
    __shared__ unsigned ldsB[128 * 20];   // [cout][kp]    stride 20 (frag = two 4-uint runs at +0/+8)

    const int tid  = threadIdx.x;
    const int lane = tid & 31;
    const int wave = tid >> 5;
    const int wm   = wave & 3;            // 16-pixel tile within 64
    const int wq   = wave >> 2;           // 64-cout half within 128

    const int HW        = H * W;
    const long p0       = (long)blockIdx.x * 64;     // global pixel over B*H*W
    const int  b        = (int)(p0 / HW);
    const int  pimg0    = (int)(p0 % HW);            // 64 consecutive pixels of image b
    const int  coutBase = blockIdx.y * 128;

    // A-staging roles: 64 pixel lanes x 4 kp-slot groups
    const int lp  = tid & 63;
    const int lc2 = tid >> 6;             // 0..3
    const int myPix = pimg0 + lp;
    const int py = myPix / W, px = myPix % W;
    const float* inBase = in + (long)b * 256 * HW;

    // B-staging roles: 128 cout rows x 2 half-rows of 8 uints (32B)
    const int bro = tid >> 1;             // 0..127 cout row
    const int bhc = (tid & 1) * 8;        // 0 or 8 (uint col within 16-kp row slice)

    const long strideIt = 8L * HW;        // A pointer step per unrolled iteration (8 channels)

    const v8f vzero = {0.f, 0.f, 0.f, 0.f, 0.f, 0.f, 0.f, 0.f};
    v8f acc[4] = { vzero, vzero, vzero, vzero };

    for (int tap = 0; tap < 9; ++tap) {
        const int dy = tap / 3 - 1, dx = tap % 3 - 1;
        const int sy = py + dy, sx = px + dx;
        const bool inb = (sy >= 0) && (sy < H) && (sx >= 0) && (sx < W);
        const long safeoff = inb ? ((long)sy * W + sx) : 0;  // clamped valid address
        const float mask = inb ? 1.f : 0.f;                  // zero-fill by multiply
        const float* pTap = inBase + safeoff;

        for (int cs = 0; cs < 8; ++cs) {                     // 32 channels per step
            __syncthreads();
            // ---- stage A: 64 px x 32 ch -> bf16 pairs, one b32 per write ----
            {
                const float* pc = pTap + (long)(cs * 32 + 2 * lc2) * HW;
                #pragma unroll
                for (int it = 0; it < 4; ++it) {
                    float v0 = pc[0] * mask;
                    float v1 = pc[HW] * mask;
                    ldsA[lp * 20 + lc2 + 4 * it] = pack_bf16x2(v0, v1);
                    pc += strideIt;
                }
            }
            // ---- stage B: 128 cout rows x 16 kp, 32B per thread (2x uint4) ----
            {
                const unsigned* sb = wpT + (size_t)(coutBase + bro) * 1152
                                         + tap * 128 + cs * 16 + bhc;
                uint4 w0 = ((const uint4*)sb)[0];
                uint4 w1 = ((const uint4*)sb)[1];
                uint4* dst = (uint4*)&ldsB[bro * 20 + bhc];
                dst[0] = w0;
                dst[1] = w1;
            }
            __builtin_prefetch(pTap + (long)(cs * 32 + 32) * HW, 0, 1);
            __syncthreads();

            // ---- fragments (documented 16-bit A 16x32 wave32 layout) ----
            const int m    = lane & 15;
            const int half = lane >> 4;
            union { unsigned u[8]; uint4 q[2]; v16bf v; } Af;
            {
                const unsigned* ar = &ldsA[(wm * 16 + m) * 20 + half * 4];
                Af.q[0] = *(const uint4*)(ar);       // kp {0..3}+4h   -> K 0..7 / 8..15
                Af.q[1] = *(const uint4*)(ar + 8);   // kp {8..11}+4h  -> K 16..23 / 24..31
            }
            #pragma unroll
            for (int j = 0; j < 4; ++j) {            // A reused across 4 N tiles
                union { unsigned u[8]; uint4 q[2]; v16bf v; } Bf;
                const unsigned* br = &ldsB[(wq * 64 + j * 16 + m) * 20 + half * 4];
                Bf.q[0] = *(const uint4*)(br);       // kp {0..3}+4h   -> K 0..7 / 8..15
                Bf.q[1] = *(const uint4*)(br + 8);   // kp {8..11}+4h  -> K 16..23 / 24..31
                acc[j] = __builtin_amdgcn_wmma_f32_16x16x32_bf16(
                             false, Af.v, false, Bf.v, (short)0, acc[j], false, false);
            }
        }
    }

    // ---- epilogue: lane holds 8 consecutive pixels -> two b128 stores/tile ----
    const int n    = lane & 15;
    const int half = lane >> 4;
    const int pimBase = pimg0 + wm * 16 + (half ? 8 : 0);
    #pragma unroll
    for (int j = 0; j < 4; ++j) {
        const int cout = coutBase + wq * 64 + j * 16 + n;
        const float bv = bias[cout];
        float* po = out + ((long)b * 256 + cout) * HW + pimBase;
        float4 lo = { acc[j][0] + bv, acc[j][1] + bv, acc[j][2] + bv, acc[j][3] + bv };
        float4 hi = { acc[j][4] + bv, acc[j][5] + bv, acc[j][6] + bv, acc[j][7] + bv };
        *(float4*)po       = lo;
        *(float4*)(po + 4) = hi;
    }
}

// ---------------------------------------------------------------------------
// Fused 2x2 maxpool + 1x1 conv (C->1) + sigmoid.
// Writes pooled features [B,256,Hp,Wp] to ws and score map [B,Hp,Wp] to d_out.
// ---------------------------------------------------------------------------
__global__ void pool_score(const float* __restrict__ conv2,
                           const float* __restrict__ wmv, const float* __restrict__ bm,
                           float* __restrict__ pooled, float* __restrict__ score,
                           int H, int W) {
    const int Hp = H >> 1, Wp = W >> 1;
    const long tot = 8L * Hp * Wp;
    long idx = (long)blockIdx.x * 256 + threadIdx.x;
    if (idx >= tot) return;
    const int b   = (int)(idx / ((long)Hp * Wp));
    const int rem = (int)(idx % ((long)Hp * Wp));
    const int pr = rem / Wp, pc = rem % Wp;
    const float* base = conv2 + (long)b * 256 * H * W;
    float dot = 0.f;
    for (int c = 0; c < 256; ++c) {
        const float* p = base + (long)c * H * W + (long)(2 * pr) * W + 2 * pc;
        float m = fmaxf(fmaxf(p[0], p[1]), fmaxf(p[W], p[W + 1]));
        pooled[(((long)b * 256 + c) * Hp + pr) * Wp + pc] = m;
        dot += m * wmv[c];
    }
    score[idx] = 1.f / (1.f + __expf(-(dot + bm[0])));
}

// ---------------------------------------------------------------------------
// Per-(level,b) top-k: iterative argmax in LDS (low-index tie-break, matches
// lax.top_k), per-selection descriptor gather + coordinate emit.
// ---------------------------------------------------------------------------
__global__ void topk_gather(const float* __restrict__ score,
                            const float* __restrict__ pooled,
                            float* __restrict__ desc, float* __restrict__ coord,
                            int Hp, int Wp, int nk, int lvl, int outOffset) {
    __shared__ float sv[4096];
    __shared__ float rv[256];
    __shared__ int   ri[256];
    const int b = blockIdx.x, t = threadIdx.x;
    const int n = Hp * Wp;
    const float* s = score + (long)b * n;
    for (int i = t; i < n; i += 256) sv[i] = s[i];
    __syncthreads();

    for (int sel = 0; sel < nk; ++sel) {
        float best = -__builtin_inff(); int bi = n;
        for (int i = t; i < n; i += 256) {
            float v = sv[i];
            if (v > best || (v == best && i < bi)) { best = v; bi = i; }
        }
        rv[t] = best; ri[t] = bi;
        __syncthreads();
        for (int st = 128; st > 0; st >>= 1) {
            if (t < st) {
                float v = rv[t + st]; int j = ri[t + st];
                if (v > rv[t] || (v == rv[t] && j < ri[t])) { rv[t] = v; ri[t] = j; }
            }
            __syncthreads();
        }
        const int idx = ri[0];
        const int r = idx / Wp, c = idx % Wp;
        const long jrow = (long)b * 448 + outOffset + sel;
        desc[jrow * 256 + t] = pooled[(((long)b * 256 + t) * Hp + r) * Wp + c];
        if (t == 0) {
            coord[jrow * 3 + 0] = (float)lvl;
            coord[jrow * 3 + 1] = (float)r;
            coord[jrow * 3 + 2] = (float)c;
            sv[idx] = -__builtin_inff();
        }
        __syncthreads();
    }
}

// ---------------------------------------------------------------------------
extern "C" void kernel_launch(void* const* d_in, const int* in_sizes, int n_in,
                              void* d_out, int out_size, void* d_ws, size_t ws_size,
                              hipStream_t stream) {
    const float* feat[3] = { (const float*)d_in[0], (const float*)d_in[1], (const float*)d_in[2] };
    const float* w1 = (const float*)d_in[3];
    const float* b1 = (const float*)d_in[4];
    const float* w2 = (const float*)d_in[5];
    const float* b2 = (const float*)d_in[6];
    const float* wm = (const float*)d_in[7];
    const float* bm = (const float*)d_in[8];
    float* out = (float*)d_out;

    // ---- workspace carve-up (bytes) ----
    char* ws = (char*)d_ws;
    const size_t WPK_BYTES = (size_t)1152 * 256 * 4;           // 1.18 MB each
    unsigned* wp1 = (unsigned*)(ws);
    unsigned* wp2 = (unsigned*)(ws + WPK_BYTES);
    float* conv1buf  = (float*)(ws + 2 * WPK_BYTES);                          // 134 MB
    float* conv2buf  = (float*)(ws + 2 * WPK_BYTES + (size_t)134217728);      // 134 MB
    float* pooledbuf = (float*)(ws + 2 * WPK_BYTES + (size_t)2 * 134217728);  // 33.5 MB

    // ---- d_out layout (floats): sm0, sm1, sm2, descriptors, coordinates ----
    float* sm[3]  = { out, out + 32768, out + 40960 };
    float* descO  = out + 43008;
    float* coordO = out + 960512;

    // ---- pack weights to bf16 K-pair layout, transposed [cout][kp] ----
    pack_w_bf16<<<1152, 256, 0, stream>>>(w1, wp1);
    pack_w_bf16<<<1152, 256, 0, stream>>>(w2, wp2);

    const int NK[3]  = { 256, 128, 64 };
    const int OFF[3] = { 0, 256, 384 };

    for (int lvl = 0; lvl < 3; ++lvl) {
        const int H = 128 >> lvl;
        const long npix = 8L * H * H;
        dim3 cgrid((unsigned)(npix / 64), 2, 1);

        conv3x3_wmma_bf16<<<cgrid, 256, 0, stream>>>(feat[lvl], wp1, b1, conv1buf, H, H);
        conv3x3_wmma_bf16<<<cgrid, 256, 0, stream>>>(conv1buf, wp2, b2, conv2buf, H, H);

        const int Hp = H >> 1;
        const long npp = 8L * Hp * Hp;
        pool_score<<<(unsigned)((npp + 255) / 256), 256, 0, stream>>>(
            conv2buf, wm, bm, pooledbuf, sm[lvl], H, H);

        topk_gather<<<8, 256, 0, stream>>>(sm[lvl], pooledbuf, descO, coordO,
                                           Hp, Hp, NK[lvl], lvl, OFF[lvl]);
    }
}
